// GptOssRouter_1176821039988
// MI455X (gfx1250) — matrix operand order repaired and checked by reading, not compiled, not run on hardware
//
#include <hip/hip_runtime.h>
#include <hip/hip_bf16.h>
#include <math.h>

// Problem constants (from reference)
#define T_TOK 16384
#define H_DIM 2880
#define E_EXP 128
#define K_TOP 4

// GEMM tiling
#define M_TILE 64          // rows per block
#define KB     64          // K-chunk per LDS stage (2880/64 = 45)
#define LDA    72          // padded LDS row stride (elements) for A
#define LDB    72          // padded LDS row stride (elements) for B
#define THREADS1 128       // 4 waves of 32

typedef __attribute__((ext_vector_type(4)))  __bf16 v4bf;
typedef __attribute__((ext_vector_type(8)))  __bf16 v8bf;
typedef __attribute__((ext_vector_type(16))) __bf16 v16bf;
typedef __attribute__((ext_vector_type(8)))  float  v8f;

static __device__ __forceinline__ v16bf cat8(v8bf lo, v8bf hi) {
    v16bf r;
#pragma unroll
    for (int i = 0; i < 8; ++i) { r[i] = lo[i]; r[i + 8] = hi[i]; }
    return r;
}

static __device__ __forceinline__ v4bf cvt4(float4 v) {
    v4bf b;
    b[0] = (__bf16)v.x; b[1] = (__bf16)v.y; b[2] = (__bf16)v.z; b[3] = (__bf16)v.w;
    return b;
}

// ---------------------------------------------------------------------------
// Kernel 0: W [E,H] f32 -> bf16 into workspace
// ---------------------------------------------------------------------------
__global__ __launch_bounds__(256)
void convert_w_kernel(const float* __restrict__ w, __bf16* __restrict__ wbf) {
    int i = (blockIdx.x * 256 + threadIdx.x) * 4;
    float4 v = *(const float4*)(w + i);
    *(v4bf*)(wbf + i) = cvt4(v);
}

// ---------------------------------------------------------------------------
// Kernel 1 (primary): logits = x @ W^T + bias.
//   A: f32 HBM -> VGPR -> bf16 -> LDS (32 staging VGPRs)
//   B: bf16 workspace -> LDS via global_load_async_to_lds_b128 (ASYNCcnt DMA)
//   Double-buffered LDS; inner schedule pinned with sched_group_barrier so
//   fragment t+1 is in flight before WMMA t issues.
// ---------------------------------------------------------------------------
__global__ __launch_bounds__(THREADS1)
void router_gemm_async(const float* __restrict__ x,
                       const __bf16* __restrict__ wbf,
                       const float* __restrict__ bias,
                       float* __restrict__ logits) {
    __shared__ __bf16 lds_a[2][M_TILE * LDA];   // 2 *  9216 B
    __shared__ __bf16 lds_b[2][E_EXP  * LDB];   // 2 * 18432 B  (total 55296 B)

    const int tid  = threadIdx.x;
    const int lane = tid & 31;
    const int wave = tid >> 5;                 // 0..3
    const int block_m0 = blockIdx.x * M_TILE;
    const int wave_m0  = wave * 16;

    const int r  = lane & 15;
    const int hi = lane >> 4;

    // A staging coords: 1024 float4s over 128 threads -> 8 each
    int a_row[8], a_k4[8];
#pragma unroll
    for (int i = 0; i < 8; ++i) {
        int f4 = tid + i * THREADS1;
        a_row[i] = f4 >> 4;
        a_k4[i]  = (f4 & 15) * 4;
    }
    // B async coords: 1024 16B-granules (128 experts * 8 granules) -> 8 each
    int b_e[8], b_k8[8];
#pragma unroll
    for (int i = 0; i < 8; ++i) {
        int g = tid + i * THREADS1;
        b_e[i]  = g >> 3;
        b_k8[i] = (g & 7) * 8;
    }

    v8f acc[8];
#pragma unroll
    for (int t = 0; t < 8; ++t)
#pragma unroll
        for (int j = 0; j < 8; ++j) acc[t][j] = 0.0f;

    float4 ra[8];

    // ---- prologue: fill buffer 0 (chunk 0) ----
#pragma unroll
    for (int i = 0; i < 8; ++i) {
        unsigned ldsoff = (unsigned)(uintptr_t)(void*)&lds_b[0][b_e[i] * LDB + b_k8[i]];
        unsigned long long ga =
            (unsigned long long)(uintptr_t)(wbf + (size_t)b_e[i] * H_DIM + b_k8[i]);
        asm volatile("global_load_async_to_lds_b128 %0, %1, off"
                     :: "v"(ldsoff), "v"(ga) : "memory");
    }
#pragma unroll
    for (int i = 0; i < 8; ++i)
        ra[i] = *(const float4*)(x + (size_t)(block_m0 + a_row[i]) * H_DIM + a_k4[i]);
#pragma unroll
    for (int i = 0; i < 8; ++i)
        *(v4bf*)&lds_a[0][a_row[i] * LDA + a_k4[i]] = cvt4(ra[i]);
    asm volatile("s_wait_asynccnt 0x0" ::: "memory");

    int parity = 0;
    for (int kc = 0; kc < H_DIM; kc += KB) {
        __syncthreads();   // publish buf[parity]; readers of buf[parity^1] done

        const int knext = kc + KB;
        const bool has_next = knext < H_DIM;
        const int pn = parity ^ 1;

        if (has_next) {
            // ---- B: async DMA next chunk straight into LDS (no VGPRs) ----
#pragma unroll
            for (int i = 0; i < 8; ++i) {
                unsigned ldsoff =
                    (unsigned)(uintptr_t)(void*)&lds_b[pn][b_e[i] * LDB + b_k8[i]];
                unsigned long long ga = (unsigned long long)(uintptr_t)
                    (wbf + (size_t)b_e[i] * H_DIM + knext + b_k8[i]);
                asm volatile("global_load_async_to_lds_b128 %0, %1, off"
                             :: "v"(ldsoff), "v"(ga) : "memory");
            }
            // ---- A: issue next chunk's f32 loads (latency overlaps WMMAs) ----
#pragma unroll
            for (int i = 0; i < 8; ++i)
                ra[i] = *(const float4*)(x + (size_t)(block_m0 + a_row[i]) * H_DIM
                                         + knext + a_k4[i]);
        }
        if (kc + 2 * KB < H_DIM) {
            __builtin_prefetch(x + (size_t)(block_m0 + a_row[0]) * H_DIM
                               + kc + 2 * KB + a_k4[0], 0, 1);
        }

        // ---- compute current chunk from buf[parity] ----
        const __bf16* brow = &lds_b[parity][r * LDB + hi * 8];
#pragma unroll
        for (int ks = 0; ks < 2; ++ks) {
            // A frag (ISA 7.12.2, 16-bit A 16x32)
            const __bf16* ap = &lds_a[parity][(wave_m0 + r) * LDA + ks * 32 + hi * 8];
            v16bf afrag = cat8(*(const v8bf*)ap, *(const v8bf*)(ap + 16));

            const __bf16* bp = brow + ks * 32;
#pragma unroll
            for (int t = 0; t < 8; ++t) {
                const __bf16* bpt = bp + t * (16 * LDB);
                v16bf bfrag = cat8(*(const v8bf*)bpt, *(const v8bf*)(bpt + 16));
                acc[t] = __builtin_amdgcn_wmma_f32_16x16x32_bf16(
                    false, afrag, false, bfrag, (short)0, acc[t], false, false);
            }

            // Pin the schedule: A(2)+B0(2)+B1(2) DS reads up front, then
            // alternate [WMMA, next-frag DS reads], tail 2 WMMAs.
            // 18 DS reads (0x100) and 8 WMMAs (0x8) per sub-chunk.
            __builtin_amdgcn_sched_group_barrier(0x100, 6, 0);
#pragma unroll
            for (int t = 0; t < 6; ++t) {
                __builtin_amdgcn_sched_group_barrier(0x008, 1, 0);
                __builtin_amdgcn_sched_group_barrier(0x100, 2, 0);
            }
            __builtin_amdgcn_sched_group_barrier(0x008, 1, 0);
            __builtin_amdgcn_sched_group_barrier(0x008, 1, 0);
        }

        // ---- stage next A; fence this iteration's async B DMA ----
        if (has_next) {
#pragma unroll
            for (int i = 0; i < 8; ++i)
                *(v4bf*)&lds_a[pn][a_row[i] * LDA + a_k4[i]] = cvt4(ra[i]);
            asm volatile("s_wait_asynccnt 0x0" ::: "memory");
        }
        parity ^= 1;
    }

    // ---- epilogue: add bias, store logits ----
    // C layout: VGPR j, lane<16 -> (M=j, N=r); lane>=16 -> (M=8+j, N=r)
    float bv[8];
#pragma unroll
    for (int t = 0; t < 8; ++t) bv[t] = bias[t * 16 + r];
#pragma unroll
    for (int t = 0; t < 8; ++t) {
#pragma unroll
        for (int j = 0; j < 8; ++j) {
            int row = block_m0 + wave_m0 + hi * 8 + j;
            int col = t * 16 + r;
            logits[(size_t)row * E_EXP + col] = acc[t][j] + bv[t];
        }
    }
}

// ---------------------------------------------------------------------------
// Kernel 1 (fallback, if ws too small): register-staged A and B
// ---------------------------------------------------------------------------
__global__ __launch_bounds__(THREADS1)
void router_gemm_fallback(const float* __restrict__ x,
                          const float* __restrict__ w,
                          const float* __restrict__ bias,
                          float* __restrict__ logits) {
    __shared__ __bf16 lds_a[2][M_TILE * LDA];
    __shared__ __bf16 lds_b[2][E_EXP  * LDB];

    const int tid  = threadIdx.x;
    const int lane = tid & 31;
    const int wave = tid >> 5;
    const int block_m0 = blockIdx.x * M_TILE;
    const int wave_m0  = wave * 16;
    const int r  = lane & 15;
    const int hi = lane >> 4;

    int a_row[8], a_k4[8], b_e[16], b_k4[16];
#pragma unroll
    for (int i = 0; i < 8; ++i) {
        int f4 = tid + i * THREADS1;
        a_row[i] = f4 >> 4; a_k4[i] = (f4 & 15) * 4;
    }
#pragma unroll
    for (int i = 0; i < 16; ++i) {
        int f4 = tid + i * THREADS1;
        b_e[i] = f4 >> 4; b_k4[i] = (f4 & 15) * 4;
    }

    v8f acc[8];
#pragma unroll
    for (int t = 0; t < 8; ++t)
#pragma unroll
        for (int j = 0; j < 8; ++j) acc[t][j] = 0.0f;

    float4 ra[8], rb[16];
#pragma unroll
    for (int i = 0; i < 8; ++i)
        ra[i] = *(const float4*)(x + (size_t)(block_m0 + a_row[i]) * H_DIM + a_k4[i]);
#pragma unroll
    for (int i = 0; i < 16; ++i)
        rb[i] = *(const float4*)(w + (size_t)b_e[i] * H_DIM + b_k4[i]);
#pragma unroll
    for (int i = 0; i < 8; ++i)
        *(v4bf*)&lds_a[0][a_row[i] * LDA + a_k4[i]] = cvt4(ra[i]);
#pragma unroll
    for (int i = 0; i < 16; ++i)
        *(v4bf*)&lds_b[0][b_e[i] * LDB + b_k4[i]] = cvt4(rb[i]);

    int parity = 0;
    for (int kc = 0; kc < H_DIM; kc += KB) {
        __syncthreads();
        const int knext = kc + KB;
        const bool has_next = knext < H_DIM;
        if (has_next) {
#pragma unroll
            for (int i = 0; i < 8; ++i)
                ra[i] = *(const float4*)(x + (size_t)(block_m0 + a_row[i]) * H_DIM + knext + a_k4[i]);
#pragma unroll
            for (int i = 0; i < 16; ++i)
                rb[i] = *(const float4*)(w + (size_t)b_e[i] * H_DIM + knext + b_k4[i]);
        }
#pragma unroll
        for (int ks = 0; ks < 2; ++ks) {
            const __bf16* ap = &lds_a[parity][(wave_m0 + r) * LDA + ks * 32 + hi * 8];
            v16bf afrag = cat8(*(const v8bf*)ap, *(const v8bf*)(ap + 16));
#pragma unroll
            for (int t = 0; t < 8; ++t) {
                const __bf16* bp = &lds_b[parity][(t * 16 + r) * LDB + ks * 32 + hi * 8];
                v16bf bfrag = cat8(*(const v8bf*)bp, *(const v8bf*)(bp + 16));
                acc[t] = __builtin_amdgcn_wmma_f32_16x16x32_bf16(
                    false, afrag, false, bfrag, (short)0, acc[t], false, false);
            }
        }
        if (has_next) {
#pragma unroll
            for (int i = 0; i < 8; ++i)
                *(v4bf*)&lds_a[parity ^ 1][a_row[i] * LDA + a_k4[i]] = cvt4(ra[i]);
#pragma unroll
            for (int i = 0; i < 16; ++i)
                *(v4bf*)&lds_b[parity ^ 1][b_e[i] * LDB + b_k4[i]] = cvt4(rb[i]);
        }
        parity ^= 1;
    }

    float bv[8];
#pragma unroll
    for (int t = 0; t < 8; ++t) bv[t] = bias[t * 16 + r];
#pragma unroll
    for (int t = 0; t < 8; ++t)
#pragma unroll
        for (int j = 0; j < 8; ++j) {
            int row = block_m0 + wave_m0 + hi * 8 + j;
            int col = t * 16 + r;
            logits[(size_t)row * E_EXP + col] = acc[t][j] + bv[t];
        }
}

// ---------------------------------------------------------------------------
// Kernel 2: per-row top-4 (desc, first-index tie-break), softmax, scatter.
// ---------------------------------------------------------------------------
__global__ __launch_bounds__(256)
void router_topk_kernel(float* __restrict__ scores, int* __restrict__ idx_out) {
    const int lane = threadIdx.x & 31;
    const int wave = threadIdx.x >> 5;
    const int row  = blockIdx.x * 8 + wave;

    float* rowp = scores + (size_t)row * E_EXP;

    float v[4];
#pragma unroll
    for (int j = 0; j < 4; ++j) v[j] = rowp[lane + j * 32];

    bool used[4] = {false, false, false, false};
    float topv[K_TOP];
    int   topi[K_TOP];

#pragma unroll
    for (int rr = 0; rr < K_TOP; ++rr) {
        float bvv = -3.402823466e38f;
        int   bii = 0x7fffffff;
#pragma unroll
        for (int j = 0; j < 4; ++j) {
            if (!used[j]) {
                float c  = v[j];
                int   ci = lane + j * 32;
                if (c > bvv || (c == bvv && ci < bii)) { bvv = c; bii = ci; }
            }
        }
#pragma unroll
        for (int off = 16; off > 0; off >>= 1) {
            float ov = __shfl_xor(bvv, off, 32);
            int   oi = __shfl_xor(bii, off, 32);
            if (ov > bvv || (ov == bvv && oi < bii)) { bvv = ov; bii = oi; }
        }
        topv[rr] = bvv;
        topi[rr] = bii;
        if ((bii & 31) == lane) used[bii >> 5] = true;
    }

    float m = topv[0];
    float p[K_TOP], s = 0.0f;
#pragma unroll
    for (int rr = 0; rr < K_TOP; ++rr) { p[rr] = __expf(topv[rr] - m); s += p[rr]; }
    float inv = 1.0f / s;
#pragma unroll
    for (int rr = 0; rr < K_TOP; ++rr) p[rr] *= inv;

#pragma unroll
    for (int j = 0; j < 4; ++j) {
        int   ci = lane + j * 32;
        float o  = 0.0f;
#pragma unroll
        for (int rr = 0; rr < K_TOP; ++rr)
            if (ci == topi[rr]) o = p[rr];
        rowp[ci] = o;
    }

    if (lane < K_TOP) idx_out[(size_t)row * K_TOP + lane] = topi[lane];
}

// ---------------------------------------------------------------------------
extern "C" void kernel_launch(void* const* d_in, const int* in_sizes, int n_in,
                              void* d_out, int out_size, void* d_ws, size_t ws_size,
                              hipStream_t stream) {
    (void)in_sizes; (void)n_in; (void)out_size;

    const float* x    = (const float*)d_in[0];   // [T, H] f32
    const float* w    = (const float*)d_in[1];   // [E, H] f32
    const float* bias = (const float*)d_in[2];   // [E]    f32

    float* scores  = (float*)d_out;                          // [T, E] f32
    int*   idx_out = (int*)(scores + (size_t)T_TOK * E_EXP); // [T, K] int32

    const size_t wbf_bytes = (size_t)E_EXP * H_DIM * sizeof(__bf16); // 737280

    if (ws_size >= wbf_bytes) {
        __bf16* wbf = (__bf16*)d_ws;
        convert_w_kernel<<<(E_EXP * H_DIM) / (256 * 4), 256, 0, stream>>>(w, wbf);
        router_gemm_async<<<T_TOK / M_TILE, THREADS1, 0, stream>>>(x, wbf, bias, scores);
    } else {
        router_gemm_fallback<<<T_TOK / M_TILE, THREADS1, 0, stream>>>(x, w, bias, scores);
    }
    router_topk_kernel<<<T_TOK / 8, 256, 0, stream>>>(scores, idx_out);
}
